// MPNNWithAttention_26027501813984
// MI455X (gfx1250) — compile-verified, matrix operand order checked by reading, not run on hardware
//
#include <hip/hip_runtime.h>
#include <hip/hip_bf16.h>
#include <math.h>

typedef __attribute__((ext_vector_type(16))) _Float16 v16h;
typedef __attribute__((ext_vector_type(8)))  float    v8f;

#define NN   50000
#define EE_  500000
#define GG   64
#define INF_ 128
#define HF   64
#define HEADS 4
#define OUTF 326000
#define NEG_SLOPE 0.2f
#define ETOT (EE_ + NN)   // edges + self loops

// ---------------- float atomic-max via order-preserving uint map ----------------
__device__ __forceinline__ unsigned f2ord(float f) {
    unsigned u = __float_as_uint(f);
    return (u & 0x80000000u) ? ~u : (u | 0x80000000u);
}
__device__ __forceinline__ float ord2f(unsigned u) {
    return (u & 0x80000000u) ? __uint_as_float(u & 0x7fffffffu) : __uint_as_float(~u);
}
#define ORD_NEG_INF 0x007FFFFFu   // f2ord(-inf)

__device__ __forceinline__ float leaky(float x) { return x > 0.0f ? x : NEG_SLOPE * x; }

// ---------------- WMMA tile loads (f32 memory -> f16 fragments) ----------------
// A: [M,K] row-major. 16x32 tile. Lane half (lane>=16) selects K-subgroup per ISA 7.12.2.
__device__ __forceinline__ v16h load_tile_a(const float* __restrict__ A, int lda,
                                            int row0, int k0, int lane) {
    int half = lane >> 4;
    const float* p = A + (size_t)(row0 + (lane & 15)) * lda + k0 + half * 8;
    float4 f0 = ((const float4*)p)[0];
    float4 f1 = ((const float4*)p)[1];
    float4 f2 = ((const float4*)(p + 16))[0];
    float4 f3 = ((const float4*)(p + 16))[1];
    v16h a;
    a[0]=(_Float16)f0.x; a[1]=(_Float16)f0.y; a[2]=(_Float16)f0.z; a[3]=(_Float16)f0.w;
    a[4]=(_Float16)f1.x; a[5]=(_Float16)f1.y; a[6]=(_Float16)f1.z; a[7]=(_Float16)f1.w;
    a[8]=(_Float16)f2.x; a[9]=(_Float16)f2.y; a[10]=(_Float16)f2.z; a[11]=(_Float16)f2.w;
    a[12]=(_Float16)f3.x; a[13]=(_Float16)f3.y; a[14]=(_Float16)f3.z; a[15]=(_Float16)f3.w;
    return a;
}
// B: [K,N] row-major; lane holds one output column, mirrored K-subgrouping.
__device__ __forceinline__ v16h load_tile_b(const float* __restrict__ B, int ldb,
                                            int k0, int col0, int lane) {
    int half = lane >> 4;
    const float* p = B + (size_t)(k0 + half * 8) * ldb + col0 + (lane & 15);
    v16h b;
#pragma unroll
    for (int i = 0; i < 8; ++i) {
        b[i]     = (_Float16)p[(size_t)i * ldb];
        b[8 + i] = (_Float16)p[(size_t)(16 + i) * ldb];
    }
    return b;
}

// ---------------- generic WMMA GEMM: C[M,Nc] = A[M,K] @ B[K,Nc] (+ bias[col]) ----------------
__global__ __launch_bounds__(128) void k_gemm_wmma(
    const float* __restrict__ A, const float* __restrict__ B,
    const float* __restrict__ bias, float* __restrict__ C,
    int M, int Nc, int K, int lda, int ldb, int ldc)
{
    int warp = threadIdx.x >> 5;
    int lane = threadIdx.x & 31;
    int ntN  = Nc >> 4;
    int tile = blockIdx.x * 4 + warp;
    if (tile >= (M >> 4) * ntN) return;       // warp-uniform: EXEC stays all-ones
    int row0 = (tile / ntN) << 4;
    int col0 = (tile % ntN) << 4;

    v8f acc = {};
    for (int k0 = 0; k0 < K; k0 += 32) {
        v16h a = load_tile_a(A, lda, row0, k0, lane);
        v16h b = load_tile_b(B, ldb, k0, col0, lane);
        acc = __builtin_amdgcn_wmma_f32_16x16x32_f16(false, a, false, b,
                                                     (short)0, acc, false, false);
    }
    int half = lane >> 4;
    int col  = col0 + (lane & 15);
    float bv = bias ? bias[col] : 0.0f;
#pragma unroll
    for (int v = 0; v < 8; ++v) {
        int row = row0 + half * 8 + v;
        C[(size_t)row * ldc + col] = acc[v] + bv;
    }
}

// ---------------- specialized final FC: out[64,OUTF] = pooled[64,64] @ W_fc + b_fc ----------
// One wave per 16-column tile; all 4 row tiles computed in-wave so each W_fc
// fragment is loaded exactly once (compulsory 83.5 MB of B traffic, no 4x re-read).
// 4 independent WMMAs per k-step also break the accumulator dependency chain.
__global__ __launch_bounds__(128) void k_fc_wmma(
    const float* __restrict__ A,    // pooled [64,64]
    const float* __restrict__ B,    // W_fc [64, OUTF]
    const float* __restrict__ bias, // b_fc [OUTF]
    float* __restrict__ C)          // out [64, OUTF]
{
    int warp = threadIdx.x >> 5;
    int lane = threadIdx.x & 31;
    int nt   = blockIdx.x * 4 + warp;
    if (nt >= OUTF / 16) return;              // warp-uniform guard
    int col0 = nt << 4;

    v8f acc0 = {}, acc1 = {}, acc2 = {}, acc3 = {};
#pragma unroll
    for (int k0 = 0; k0 < HF; k0 += 32) {
        v16h b  = load_tile_b(B, OUTF, k0, col0, lane);   // loaded ONCE per k-step
        v16h a0 = load_tile_a(A, HF,  0, k0, lane);
        v16h a1 = load_tile_a(A, HF, 16, k0, lane);
        v16h a2 = load_tile_a(A, HF, 32, k0, lane);
        v16h a3 = load_tile_a(A, HF, 48, k0, lane);
        acc0 = __builtin_amdgcn_wmma_f32_16x16x32_f16(false, a0, false, b, (short)0, acc0, false, false);
        acc1 = __builtin_amdgcn_wmma_f32_16x16x32_f16(false, a1, false, b, (short)0, acc1, false, false);
        acc2 = __builtin_amdgcn_wmma_f32_16x16x32_f16(false, a2, false, b, (short)0, acc2, false, false);
        acc3 = __builtin_amdgcn_wmma_f32_16x16x32_f16(false, a3, false, b, (short)0, acc3, false, false);
    }
    int half = lane >> 4;
    int col  = col0 + (lane & 15);
    float bv = bias[col];
#pragma unroll
    for (int v = 0; v < 8; ++v) {
        int r = half * 8 + v;
        C[(size_t)(r)      * OUTF + col] = acc0[v] + bv;
        C[(size_t)(r + 16) * OUTF + col] = acc1[v] + bv;
        C[(size_t)(r + 32) * OUTF + col] = acc2[v] + bv;
        C[(size_t)(r + 48) * OUTF + col] = acc3[v] + bv;
    }
}

// ---------------- attention logits: ls/ld[n][h] = <xt[n,h,:], a_src/dst[h,:]> ----------------
__global__ void k_logits(const float* __restrict__ xt, const float* __restrict__ asrc,
                         const float* __restrict__ adst, float* __restrict__ ls,
                         float* __restrict__ ld)
{
    int i = blockIdx.x * blockDim.x + threadIdx.x;   // over N*HEADS
    if (i >= NN * HEADS) return;
    int n = i >> 2, h = i & 3;
    const float* xp = xt + (size_t)n * (HEADS * HF) + h * HF;
    const float* as = asrc + h * HF;
    const float* ad = adst + h * HF;
    float s = 0.0f, d = 0.0f;
#pragma unroll 8
    for (int c = 0; c < HF; ++c) { float x = xp[c]; s += x * as[c]; d += x * ad[c]; }
    ls[i] = s; ld[i] = d;
}

// ---------------- per-layer init: agg=0, denom=0, m=ord(-inf) ----------------
__global__ void k_layer_init(float* __restrict__ agg, float* __restrict__ denom,
                             unsigned* __restrict__ mo)
{
    int i = blockIdx.x * blockDim.x + threadIdx.x;   // over N*HEADS*HF
    if (i >= NN * HEADS * HF) return;
    agg[i] = 0.0f;
    if (i < NN * HEADS) { denom[i] = 0.0f; mo[i] = ORD_NEG_INF; }
}

// ---------------- edge segment-max ----------------
__global__ void k_edge_max(const int* __restrict__ src, const int* __restrict__ dst,
                           const float* __restrict__ ls, const float* __restrict__ ld,
                           unsigned* __restrict__ mo)
{
    int e = blockIdx.x * blockDim.x + threadIdx.x;
    if (e >= ETOT) return;
    int s = (e < EE_) ? src[e] : (e - EE_);
    int d = (e < EE_) ? dst[e] : (e - EE_);
#pragma unroll
    for (int h = 0; h < HEADS; ++h) {
        float v = leaky(ls[s * HEADS + h] + ld[d * HEADS + h]);
        atomicMax(&mo[d * HEADS + h], f2ord(v));
    }
}

// m_ord(uint) -> m(float) in place, non-finite -> 0 (reference semantics)
__global__ void k_fix_m(unsigned* __restrict__ mo, float* __restrict__ mf)
{
    int i = blockIdx.x * blockDim.x + threadIdx.x;
    if (i >= NN * HEADS) return;
    float v = ord2f(mo[i]);
    mf[i] = isfinite(v) ? v : 0.0f;
}

// ---------------- edge segment-sum of exp ----------------
__global__ void k_edge_sum(const int* __restrict__ src, const int* __restrict__ dst,
                           const float* __restrict__ ls, const float* __restrict__ ld,
                           const float* __restrict__ mf, float* __restrict__ denom)
{
    int e = blockIdx.x * blockDim.x + threadIdx.x;
    if (e >= ETOT) return;
    int s = (e < EE_) ? src[e] : (e - EE_);
    int d = (e < EE_) ? dst[e] : (e - EE_);
#pragma unroll
    for (int h = 0; h < HEADS; ++h) {
        float v = leaky(ls[s * HEADS + h] + ld[d * HEADS + h]);
        atomicAdd(&denom[d * HEADS + h], __expf(v - mf[d * HEADS + h]));
    }
}

// ---------------- edge aggregation: agg[dst,h,:] += alpha * xt[src,h,:] ----------------
// 64 threads per edge (one per channel), 4 edges per 256-thread block.
__global__ __launch_bounds__(256) void k_edge_agg(
    const int* __restrict__ src, const int* __restrict__ dst,
    const float* __restrict__ ls, const float* __restrict__ ld,
    const float* __restrict__ mf, const float* __restrict__ denom,
    const float* __restrict__ xt, float* __restrict__ agg)
{
    int le = threadIdx.x >> 6;           // 0..3
    int c  = threadIdx.x & 63;           // channel
    int e  = blockIdx.x * 4 + le;
    if (e >= ETOT) return;
    int s = (e < EE_) ? src[e] : (e - EE_);
    int d = (e < EE_) ? dst[e] : (e - EE_);
#pragma unroll
    for (int h = 0; h < HEADS; ++h) {
        float v  = leaky(ls[s * HEADS + h] + ld[d * HEADS + h]);
        float ex = __expf(v - mf[d * HEADS + h]);
        float alpha = ex / (denom[d * HEADS + h] + 1e-16f);
        float val = xt[(size_t)s * (HEADS * HF) + h * HF + c] * alpha;
        atomicAdd(&agg[(size_t)d * (HEADS * HF) + h * HF + c], val);
    }
}

// ---------------- head mean + bias + relu ----------------
__global__ void k_head_mean(const float* __restrict__ agg, const float* __restrict__ b,
                            float* __restrict__ hnext)
{
    int i = blockIdx.x * blockDim.x + threadIdx.x;   // over N*HF
    if (i >= NN * HF) return;
    int n = i >> 6, c = i & 63;
    const float* a = agg + (size_t)n * (HEADS * HF) + c;
    float s = (a[0] + a[HF] + a[2 * HF] + a[3 * HF]) * 0.25f + b[c];
    hnext[i] = s > 0.0f ? s : 0.0f;
}

// ---------------- pooling ----------------
__global__ void k_pool_zero(float* __restrict__ pooled, float* __restrict__ cnt)
{
    int i = blockIdx.x * blockDim.x + threadIdx.x;
    if (i < GG * HF) pooled[i] = 0.0f;
    if (i < GG) cnt[i] = 0.0f;
}
__global__ void k_pool_sum(const float* __restrict__ h, const int* __restrict__ batch,
                           float* __restrict__ pooled, float* __restrict__ cnt)
{
    int i = blockIdx.x * blockDim.x + threadIdx.x;   // over N*HF
    if (i >= NN * HF) return;
    int n = i >> 6, c = i & 63;
    int g = batch[n];
    atomicAdd(&pooled[g * HF + c], h[i]);
    if (c == 0) atomicAdd(&cnt[g], 1.0f);
}
__global__ void k_pool_div(float* __restrict__ pooled, const float* __restrict__ cnt)
{
    int i = blockIdx.x * blockDim.x + threadIdx.x;
    if (i >= GG * HF) return;
    float c = cnt[i >> 6];
    pooled[i] /= fmaxf(c, 1.0f);
}

static inline int cdiv(long long a, long long b) { return (int)((a + b - 1) / b); }

extern "C" void kernel_launch(void* const* d_in, const int* in_sizes, int n_in,
                              void* d_out, int out_size, void* d_ws, size_t ws_size,
                              hipStream_t stream) {
    const float* x      = (const float*)d_in[0];
    const int*   ei     = (const int*)d_in[1];
    const int*   batch  = (const int*)d_in[3];
    const float* W_emb  = (const float*)d_in[4];
    const float* b_emb  = (const float*)d_in[5];
    const float* W_fc   = (const float*)d_in[6];
    const float* b_fc   = (const float*)d_in[7];
    const float* Wl[3]   = { (const float*)d_in[8],  (const float*)d_in[12], (const float*)d_in[16] };
    const float* asl[3]  = { (const float*)d_in[9],  (const float*)d_in[13], (const float*)d_in[17] };
    const float* adl[3]  = { (const float*)d_in[10], (const float*)d_in[14], (const float*)d_in[18] };
    const float* bl[3]   = { (const float*)d_in[11], (const float*)d_in[15], (const float*)d_in[19] };

    const int* srcp = ei;
    const int* dstp = ei + EE_;

    // workspace layout (floats)
    float* w = (float*)d_ws;
    float* h0     = w;                      w += (size_t)NN * HF;           // 12.8 MB
    float* h1     = w;                      w += (size_t)NN * HF;
    float* xt     = w;                      w += (size_t)NN * HEADS * HF;   // 51.2 MB
    float* ls     = w;                      w += (size_t)NN * HEADS;
    float* ld     = w;                      w += (size_t)NN * HEADS;
    float* mbuf   = w;                      w += (size_t)NN * HEADS;        // uint during max
    float* denom  = w;                      w += (size_t)NN * HEADS;
    float* agg    = w;                      w += (size_t)NN * HEADS * HF;   // 51.2 MB
    float* pooled = w;                      w += (size_t)GG * HF;
    float* cnt    = w;                      w += (size_t)GG;
    unsigned* mo  = (unsigned*)mbuf;

    // 1) embedding: h0 = x @ W_emb + b_emb   (M=50000, Nc=64, K=128)
    {
        int tiles = (NN / 16) * (HF / 16);
        k_gemm_wmma<<<cdiv(tiles, 4), 128, 0, stream>>>(x, W_emb, b_emb, h0,
                                                        NN, HF, INF_, INF_, HF, HF);
    }

    float* hin = h0, *hout = h1;
    for (int l = 0; l < 3; ++l) {
        // xt = hin @ Wl  (M=50000, Nc=256, K=64)
        int tiles = (NN / 16) * ((HEADS * HF) / 16);
        k_gemm_wmma<<<cdiv(tiles, 4), 128, 0, stream>>>(hin, Wl[l], nullptr, xt,
                                                        NN, HEADS * HF, HF, HF, HEADS * HF, HEADS * HF);
        k_logits<<<cdiv((long long)NN * HEADS, 256), 256, 0, stream>>>(xt, asl[l], adl[l], ls, ld);
        k_layer_init<<<cdiv((long long)NN * HEADS * HF, 256), 256, 0, stream>>>(agg, denom, mo);
        k_edge_max<<<cdiv(ETOT, 256), 256, 0, stream>>>(srcp, dstp, ls, ld, mo);
        k_fix_m<<<cdiv((long long)NN * HEADS, 256), 256, 0, stream>>>(mo, mbuf);
        k_edge_sum<<<cdiv(ETOT, 256), 256, 0, stream>>>(srcp, dstp, ls, ld, mbuf, denom);
        k_edge_agg<<<cdiv(ETOT, 4), 256, 0, stream>>>(srcp, dstp, ls, ld, mbuf, denom, xt, agg);
        k_head_mean<<<cdiv((long long)NN * HF, 256), 256, 0, stream>>>(agg, bl[l], hout);
        float* t = hin; hin = hout; hout = t;
    }

    // pooling over graphs
    k_pool_zero<<<cdiv(GG * HF, 256), 256, 0, stream>>>(pooled, cnt);
    k_pool_sum<<<cdiv((long long)NN * HF, 256), 256, 0, stream>>>(hin, batch, pooled, cnt);
    k_pool_div<<<cdiv(GG * HF, 256), 256, 0, stream>>>(pooled, cnt);

    // final FC: out[64, 326000] = pooled @ W_fc + b_fc  (M=64 all in-wave, Nc=326000, K=64)
    k_fc_wmma<<<cdiv(OUTF / 16, 4), 128, 0, stream>>>(pooled, W_fc, b_fc, (float*)d_out);

    (void)in_sizes; (void)n_in; (void)out_size; (void)ws_size;
}
